// BertSelfAttention_81913616269449
// MI455X (gfx1250) — matrix operand order repaired and checked by reading, not compiled, not run on hardware
//
#include <hip/hip_runtime.h>

// BERT self-attention, fused flash-style, bf16 WMMA on gfx1250 (MI455X).
// B=2, S=2048, D=1024, H=16, HD=64.
// Round 2: 16x64 GEMM strips (4x A reuse), attention K/V staged into LDS
// via global_load_async_to_lds_b128 (ASYNCcnt) with double buffering.

#define BB 2
#define SS 2048
#define DD 1024
#define HH 16
#define HDIM 64

typedef __attribute__((ext_vector_type(8)))  float          v8f;
typedef __attribute__((ext_vector_type(16))) __bf16         v16bf;
typedef __attribute__((ext_vector_type(8)))  unsigned short ushort8;
typedef __attribute__((ext_vector_type(4)))  unsigned short ushort4v;

union AFrag {
    v16bf   v;
    ushort8 h[2];
};

__device__ __forceinline__ unsigned short f2bf(float f) {
    unsigned int u = __float_as_uint(f);
    u += 0x7FFFu + ((u >> 16) & 1u);          // round-to-nearest-even
    return (unsigned short)(u >> 16);
}

// LDS aperture: generic shared-memory pointer's low 32 bits are the
// wave-relative LDS byte offset (ISA 10.2).
__device__ __forceinline__ unsigned lds_off(const void* p) {
    return (unsigned)(size_t)p;
}

// Async global->LDS copy of 16 bytes (ASYNCcnt-tracked, GVS mode).
__device__ __forceinline__ void async_cp16(unsigned lds, const void* base,
                                           unsigned byte_off) {
    asm volatile("global_load_async_to_lds_b128 %0, %1, %2"
                 :: "v"(lds), "v"(byte_off), "s"(base) : "memory");
}

// ---------------------------------------------------------------- kernel 1
__global__ __launch_bounds__(256) void cvt_x_kernel(
    const float* __restrict__ x, unsigned short* __restrict__ o, int n) {
    int i = (blockIdx.x * 256 + threadIdx.x) * 4;
    if (i >= n) return;
    float4 f = *(const float4*)(x + i);
    ushort4v r;
    r[0] = f2bf(f.x); r[1] = f2bf(f.y); r[2] = f2bf(f.z); r[3] = f2bf(f.w);
    *(ushort4v*)(o + i) = r;
}

// ---------------------------------------------------------------- kernel 2
// fp32 -> bf16 + transpose of the three weight matrices: Wt[n][k] = W[k][n].
__global__ __launch_bounds__(256) void cvt_w_kernel(
    const float* __restrict__ wq, const float* __restrict__ wk,
    const float* __restrict__ wv, unsigned short* __restrict__ outbase) {
    int idx = blockIdx.x * 256 + threadIdx.x;       // 0 .. D*D-1
    int w   = blockIdx.y;                           // 0,1,2
    const float* src = (w == 0) ? wq : (w == 1) ? wk : wv;
    unsigned short* dst = outbase + (size_t)w * DD * DD;
    int k = idx >> 10;
    int n = idx & (DD - 1);
    dst[n * DD + k] = f2bf(src[idx]);
}

// ---------------------------------------------------------------- kernel 3
// QKV projection GEMM. One wave per 16x64 output strip: the A-fragment is
// loaded once per k-step and reused by 4 WMMAs (4 N-tiles).
// Q,K stored [B,H,S,HD] bf16; V stored transposed [B,H,HD,S] bf16.
__global__ __launch_bounds__(256) void qkv_gemm_kernel(
    const unsigned short* __restrict__ xbf,
    const unsigned short* __restrict__ wt,
    const float* __restrict__ bq, const float* __restrict__ bk,
    const float* __restrict__ bv,
    unsigned short* __restrict__ qbf, unsigned short* __restrict__ kbf,
    unsigned short* __restrict__ vtb) {
    int lane = threadIdx.x & 31;
    int wid  = blockIdx.x * 8 + (threadIdx.x >> 5);
    int mat  = wid >> 12;                 // 4096 strips per matrix (256 x 16)
    int t    = wid & 4095;
    int mt   = t >> 4;                    // 0..255  (M tile)
    int nt4  = t & 15;                    // 0..15   (group of 4 N tiles)
    int l16  = lane & 15;
    int hi   = lane >> 4;

    const unsigned short* w    = wt + (size_t)mat * DD * DD;
    const float*          bias = (mat == 0) ? bq : (mat == 1) ? bk : bv;

    int col[4];
    v8f acc[4];
#pragma unroll
    for (int j = 0; j < 4; ++j) {
        col[j] = nt4 * 64 + j * 16 + l16;
        float b0 = bias[col[j]];
#pragma unroll
        for (int r = 0; r < 8; ++r) acc[j][r] = b0;
    }

    int arow = mt * 16 + l16;
    const unsigned short* aptr = xbf + (size_t)arow * DD + (hi ? 16 : 0);
    const unsigned short* bptr[4];
#pragma unroll
    for (int j = 0; j < 4; ++j)
        bptr[j] = w + (size_t)col[j] * DD + (hi ? 16 : 0);

    for (int k0 = 0; k0 < DD; k0 += 32) {
        AFrag a;
        a.h[0] = *(const ushort8*)(aptr + k0);        // K 0-7  / 16-23
        a.h[1] = *(const ushort8*)(aptr + k0 + 8);    // K 8-15 / 24-31
#pragma unroll
        for (int j = 0; j < 4; ++j) {
            AFrag b;
            b.h[0] = *(const ushort8*)(bptr[j] + k0);   // K 0-15 / 16-31
            b.h[1] = *(const ushort8*)(bptr[j] + k0 + 8);
            acc[j] = __builtin_amdgcn_wmma_f32_16x16x32_bf16(
                false, a.v, false, b.v, (short)0, acc[j], false, false);
        }
    }

#pragma unroll
    for (int j = 0; j < 4; ++j) {
        int hh = col[j] >> 6;
        int hd = col[j] & (HDIM - 1);
#pragma unroll
        for (int r = 0; r < 8; ++r) {
            int grow = mt * 16 + r + hi * 8;
            int bi = grow >> 11;
            int si = grow & (SS - 1);
            unsigned short val = f2bf(acc[j][r]);
            if (mat == 0)
                qbf[(((size_t)(bi * HH + hh)) * SS + si) * HDIM + hd] = val;
            else if (mat == 1)
                kbf[(((size_t)(bi * HH + hh)) * SS + si) * HDIM + hd] = val;
            else
                vtb[(((size_t)(bi * HH + hh)) * HDIM + hd) * SS + si] = val;
        }
    }
}

// ---------------------------------------------------------------- kernel 4
// Fused flash attention. 8 waves per block share (b,h): K/V tiles are
// async-copied to LDS once per block (double-buffered), WMMA B-fragments
// come from LDS. One wave per 16-row q tile.
__global__ __launch_bounds__(256) void attn_kernel(
    const unsigned short* __restrict__ qbf,
    const unsigned short* __restrict__ kbf,
    const unsigned short* __restrict__ vtb,
    const float* __restrict__ mask, float* __restrict__ out) {
    __shared__ __align__(16) unsigned short kbuf[2][32 * HDIM];  // [key][hd]
    __shared__ __align__(16) unsigned short vbuf[2][HDIM * 32];  // [hd][key]
    __shared__ __align__(16) unsigned short plds[8 * 512];       // P transpose

    int tid  = threadIdx.x;
    int wave = tid >> 5;
    int lane = tid & 31;
    int l16  = lane & 15;
    int hi   = lane >> 4;
    int wid  = blockIdx.x * 8 + wave;
    int b    = wid >> 11;
    int h    = (wid >> 7) & 15;
    int qt   = wid & 127;
    unsigned short* my = &plds[wave * 512];

    const unsigned short* qb = qbf + (size_t)(b * HH + h) * SS * HDIM;
    const unsigned short* kb = kbf + (size_t)(b * HH + h) * SS * HDIM;
    const unsigned short* vb = vtb + (size_t)(b * HH + h) * HDIM * SS;

    // Per-thread async-copy slices: K tile 32x64 bf16 (4KB), V tile 64x32 (4KB),
    // 256 threads x 16B each.
    int krow = tid >> 3, koff = (tid & 7) * 8;    // key row, ushort offset
    int vrow = tid >> 2, voff = (tid & 3) * 8;    // hd row, ushort offset
    unsigned klds[2], vlds[2];
#pragma unroll
    for (int p = 0; p < 2; ++p) {
        klds[p] = lds_off(&kbuf[p][krow * HDIM + koff]);
        vlds[p] = lds_off(&vbuf[p][vrow * 32 + voff]);
    }

    // Persistent Q A-fragments (K = 0..31 and 32..63 of head dim).
    AFrag aq0, aq1;
    {
        const unsigned short* qp = qb + (size_t)(qt * 16 + l16) * HDIM + (hi ? 16 : 0);
        aq0.h[0] = *(const ushort8*)(qp);
        aq0.h[1] = *(const ushort8*)(qp + 8);
        aq1.h[0] = *(const ushort8*)(qp + 32);
        aq1.h[1] = *(const ushort8*)(qp + 40);
    }

    float m[8], l[8];
#pragma unroll
    for (int r = 0; r < 8; ++r) { m[r] = -3.0e38f; l[r] = 0.f; }
    v8f o[4];
#pragma unroll
    for (int tt = 0; tt < 4; ++tt)
#pragma unroll
        for (int r = 0; r < 8; ++r) o[tt][r] = 0.f;

    const float* mrow = mask + (size_t)b * SS;

    // Prefetch tile 0.
    async_cp16(klds[0], kb, (unsigned)(((0 + krow) * HDIM + koff) * 2));
    async_cp16(vlds[0], vb, (unsigned)((vrow * SS + 0 + voff) * 2));
    asm volatile("s_wait_asynccnt 0x0" ::: "memory");
    __syncthreads();

    for (int kt2 = 0; kt2 < SS / 32; ++kt2) {
        int c0  = kt2 * 32;
        int cur = kt2 & 1;
        // Prefetch next tile into the other buffer (overlaps compute).
        if (kt2 + 1 < SS / 32) {
            int nc0 = c0 + 32;
            async_cp16(klds[cur ^ 1], kb, (unsigned)(((nc0 + krow) * HDIM + koff) * 2));
            async_cp16(vlds[cur ^ 1], vb, (unsigned)((vrow * SS + nc0 + voff) * 2));
        }

        v8f sc[2];
#pragma unroll
        for (int j = 0; j < 2; ++j) {
            int colk = j * 16 + l16;                 // key index within tile
            const unsigned short* kp = &kbuf[cur][colk * HDIM + (hi ? 16 : 0)];
            AFrag bk0, bk1;
            bk0.h[0] = *(const ushort8*)(kp);
            bk0.h[1] = *(const ushort8*)(kp + 8);
            bk1.h[0] = *(const ushort8*)(kp + 32);
            bk1.h[1] = *(const ushort8*)(kp + 40);
            v8f s0;
#pragma unroll
            for (int r = 0; r < 8; ++r) s0[r] = 0.f;
            s0 = __builtin_amdgcn_wmma_f32_16x16x32_bf16(
                false, aq0.v, false, bk0.v, (short)0, s0, false, false);
            s0 = __builtin_amdgcn_wmma_f32_16x16x32_bf16(
                false, aq1.v, false, bk1.v, (short)0, s0, false, false);
            float mv = mrow[c0 + colk];
#pragma unroll
            for (int r = 0; r < 8; ++r) sc[j][r] = s0[r] * 0.125f + mv;
        }

        // Online softmax (rows live in a 16-lane half-group).
#pragma unroll
        for (int r = 0; r < 8; ++r) {
            float v = fmaxf(sc[0][r], sc[1][r]);
            v = fmaxf(v, __shfl_xor(v, 1, 32));
            v = fmaxf(v, __shfl_xor(v, 2, 32));
            v = fmaxf(v, __shfl_xor(v, 4, 32));
            v = fmaxf(v, __shfl_xor(v, 8, 32));
            float mn    = fmaxf(m[r], v);
            float alpha = __expf(m[r] - mn);
            m[r] = mn;
            float p0 = __expf(sc[0][r] - mn);
            float p1 = __expf(sc[1][r] - mn);
            sc[0][r] = p0;
            sc[1][r] = p1;
            float rs = p0 + p1;
            rs += __shfl_xor(rs, 1, 32);
            rs += __shfl_xor(rs, 2, 32);
            rs += __shfl_xor(rs, 4, 32);
            rs += __shfl_xor(rs, 8, 32);
            l[r] = l[r] * alpha + rs;
            o[0][r] *= alpha; o[1][r] *= alpha;
            o[2][r] *= alpha; o[3][r] *= alpha;
        }

        // C-layout -> A-layout transpose of P through wave-private LDS.
        int prow = hi * 8;
#pragma unroll
        for (int r = 0; r < 8; ++r) {
            my[(r + prow) * 32 + l16]      = f2bf(sc[0][r]);
            my[(r + prow) * 32 + 16 + l16] = f2bf(sc[1][r]);
        }
        asm volatile("s_wait_dscnt 0x0" ::: "memory");
        AFrag ap;
        {
            const unsigned short* lp = my + l16 * 32 + (hi ? 16 : 0);
            ap.h[0] = *(const ushort8*)(lp);
            ap.h[1] = *(const ushort8*)(lp + 8);
        }

        // O += P * V  (V tile in LDS, [hd][key]).
#pragma unroll
        for (int tt = 0; tt < 4; ++tt) {
            const unsigned short* vp = &vbuf[cur][(tt * 16 + l16) * 32 + (hi ? 16 : 0)];
            AFrag bvf;
            bvf.h[0] = *(const ushort8*)(vp);
            bvf.h[1] = *(const ushort8*)(vp + 8);
            o[tt] = __builtin_amdgcn_wmma_f32_16x16x32_bf16(
                false, ap.v, false, bvf.v, (short)0, o[tt], false, false);
        }

        // Wave's prefetch must land, then block-wide handoff.
        asm volatile("s_wait_asynccnt 0x0" ::: "memory");
        __syncthreads();
    }

#pragma unroll
    for (int r = 0; r < 8; ++r) l[r] = 1.0f / l[r];
    int srow0 = qt * 16 + hi * 8;
#pragma unroll
    for (int tt = 0; tt < 4; ++tt)
#pragma unroll
        for (int r = 0; r < 8; ++r) {
            int si = srow0 + r;
            int di = h * HDIM + tt * 16 + l16;
            out[((size_t)b * SS + si) * DD + di] = o[tt][r] * l[r];
        }
}

// ---------------------------------------------------------------- launch
extern "C" void kernel_launch(void* const* d_in, const int* in_sizes, int n_in,
                              void* d_out, int out_size, void* d_ws, size_t ws_size,
                              hipStream_t stream) {
    const float* hs   = (const float*)d_in[0];
    const float* mask = (const float*)d_in[1];
    const float* Wq   = (const float*)d_in[2];
    const float* bq   = (const float*)d_in[3];
    const float* Wk   = (const float*)d_in[4];
    const float* bk   = (const float*)d_in[5];
    const float* Wv   = (const float*)d_in[6];
    const float* bv   = (const float*)d_in[7];
    float* out        = (float*)d_out;

    char* ws = (char*)d_ws;
    unsigned short* xbf = (unsigned short*)(ws);                          // 8 MB
    unsigned short* wt  = (unsigned short*)(ws + (8u << 20));             // 6 MB
    unsigned short* qbf = (unsigned short*)(ws + (14u << 20));            // 8 MB
    unsigned short* kbf = (unsigned short*)(ws + (22u << 20));            // 8 MB
    unsigned short* vtb = (unsigned short*)(ws + (30u << 20));            // 8 MB

    (void)in_sizes; (void)n_in; (void)out_size; (void)ws_size;

    cvt_x_kernel<<<4096, 256, 0, stream>>>(hs, xbf, BB * SS * DD);
    dim3 gw(4096, 3);
    cvt_w_kernel<<<gw, 256, 0, stream>>>(Wq, Wk, Wv, wt);
    qkv_gemm_kernel<<<1536, 256, 0, stream>>>(xbf, wt, bq, bk, bv, qbf, kbf, vtb);
    attn_kernel<<<512, 256, 0, stream>>>(qbf, kbf, vtb, mask, out);
}
